// PointNetFeaturePropagation_40785009443185
// MI455X (gfx1250) — matrix operand order repaired and checked by reading, not compiled, not run on hardware
//
#include <hip/hip_runtime.h>

// ---------------------------------------------------------------------------
// PointNet Feature Propagation for MI455X (gfx1250, wave32, WMMA)
//
// Sizes (fixed by the reference):
//   B=2, N=16384, S=4096, Ds=128, Dt=256, Cin=384, C0=256, C1=128, M=B*N=32768
//
// Roofline: ~10 GFLOP total vs ~150 MB of traffic -> bandwidth bound at
// 23.3 TB/s.  GEMMs use v_wmma_f32_16x16x32_f16 (f16 in, fp32 accumulate)
// with 4-way N register blocking and explicit double buffering: one
// A-fragment load feeds 4 WMMAs, next k-step operands are in flight while
// the current WMMAs execute.  f16 staging halves intermediate bytes;
// weights (<0.5 MB) are L2 resident.  Conv bias cancels exactly under
// training-mode BN and is skipped.
// ---------------------------------------------------------------------------

typedef __attribute__((ext_vector_type(16))) _Float16 v16h;
typedef __attribute__((ext_vector_type(8)))  _Float16 v8h;
typedef __attribute__((ext_vector_type(8)))  float    v8f;

#define BB   2
#define NPT  16384
#define SPT  4096
#define DSF  128
#define DTF  256
#define CIN  384
#define C0F  256
#define C1F  128
#define MROW (BB * NPT)          // 32768 rows
#define KNN_TILE 512

// ---------------- workspace layout (bytes) ----------------
#define OFF_FEAT  ((size_t)0)                       // f16 [MROW][CIN]   = 24 MB
#define OFF_WB0   ((size_t)25165824)                // f16 [C0][CIN]
#define OFF_WB1   ((size_t)25362432)                // f16 [C1][C0]
#define OFF_X0    ((size_t)25427968)                // f32 [MROW][C0]    = 32 MB
#define OFF_A1    ((size_t)58982400)                // f16 [MROW][C0]    = 16 MB
#define OFF_IDX   ((size_t)75759616)                // i32 [MROW][3]
#define OFF_WTS   ((size_t)76152832)                // f32 [MROW][3]
#define OFF_STATS ((size_t)76546048)                // f32 scratch below
// float-indexed layout inside STATS region:
#define ST_PSUM0   0        // [128*256]
#define ST_PSQ0    32768    // [128*256]
#define ST_PSUM1   65536    // [128*128]
#define ST_PSQ1    81920    // [128*128]
#define ST_SCALE0  98304    // [256]
#define ST_SHIFT0  98560    // [256]
#define ST_SCALE1  98816    // [128]
#define ST_SHIFT1  98944    // [128]

// ---------------------------------------------------------------------------
// Convert fp32 weights to f16 (kept in (O, C) row-major so that a WMMA
// B-fragment lane load is a contiguous 16/32-byte run).
// ---------------------------------------------------------------------------
__global__ void cvt_weights_kernel(const float* __restrict__ w0,
                                   const float* __restrict__ w1,
                                   _Float16* __restrict__ wb0,
                                   _Float16* __restrict__ wb1) {
  int e = blockIdx.x * 256 + threadIdx.x;          // 512 blocks * 256
  const int n0 = C0F * CIN;                        // 98304
  if (e < n0)                 wb0[e]      = (_Float16)w0[e];
  else                        wb1[e - n0] = (_Float16)w1[e - n0];
}

// ---------------------------------------------------------------------------
// Brute-force 3-NN.  One thread per query point; xyz_t tiles staged in LDS.
// Writes flat feats_t row indices (b*S + s) and normalized inverse-distance
// weights.  Strict '<' keeps the lowest index on ties, like lax.top_k.
// ---------------------------------------------------------------------------
__global__ void knn_kernel(const float* __restrict__ xyz_s,
                           const float* __restrict__ xyz_t,
                           int*   __restrict__ idx,
                           float* __restrict__ wts) {
  __shared__ float lt[KNN_TILE * 3];
  const int p = blockIdx.x * 256 + threadIdx.x;    // 128 blocks * 256
  const int b = p >> 14;                           // p / N
  const float sx = xyz_s[p * 3 + 0];
  const float sy = xyz_s[p * 3 + 1];
  const float sz = xyz_s[p * 3 + 2];
  const float* tb = xyz_t + (size_t)b * SPT * 3;

  float bd0 = 3.0e38f, bd1 = 3.0e38f, bd2 = 3.0e38f;
  int   bi0 = 0,       bi1 = 0,       bi2 = 0;

  for (int s0 = 0; s0 < SPT; s0 += KNN_TILE) {
    for (int i = threadIdx.x; i < KNN_TILE * 3; i += 256)
      lt[i] = tb[s0 * 3 + i];
    __syncthreads();
    #pragma unroll 4
    for (int j = 0; j < KNN_TILE; ++j) {
      const float dx = sx - lt[3 * j + 0];
      const float dy = sy - lt[3 * j + 1];
      const float dz = sz - lt[3 * j + 2];
      const float d2 = dx * dx + dy * dy + dz * dz;
      const int   s  = s0 + j;
      if (d2 < bd2) {
        if (d2 < bd1) {
          if (d2 < bd0) { bd2=bd1; bi2=bi1; bd1=bd0; bi1=bi0; bd0=d2; bi0=s; }
          else          { bd2=bd1; bi2=bi1; bd1=d2;  bi1=s; }
        } else          { bd2=d2;  bi2=s; }
      }
    }
    __syncthreads();
  }
  const float r0 = 1.0f / (bd0 + 1e-8f);
  const float r1 = 1.0f / (bd1 + 1e-8f);
  const float r2 = 1.0f / (bd2 + 1e-8f);
  const float rs = 1.0f / (r0 + r1 + r2);
  wts[p * 3 + 0] = r0 * rs;
  wts[p * 3 + 1] = r1 * rs;
  wts[p * 3 + 2] = r2 * rs;
  idx[p * 3 + 0] = b * SPT + bi0;
  idx[p * 3 + 1] = b * SPT + bi1;
  idx[p * 3 + 2] = b * SPT + bi2;
}

// ---------------------------------------------------------------------------
// Fused gather + weighted interpolation + concat -> f16 feature matrix
// [MROW][CIN] (skip 0..127 | interpolated 128..383).  One block per point,
// channel-parallel threads -> coalesced feats_t row reads.
// ---------------------------------------------------------------------------
__global__ void build_features_kernel(const float* __restrict__ feats_s,
                                      const float* __restrict__ feats_t,
                                      const int*   __restrict__ idx,
                                      const float* __restrict__ wts,
                                      _Float16*    __restrict__ featA) {
  const int p = blockIdx.x;                        // MROW blocks, 128 threads
  const int c = threadIdx.x;
  featA[(size_t)p * CIN + c] = (_Float16)feats_s[(size_t)p * DSF + c];
  const int   i0 = idx[p * 3 + 0], i1 = idx[p * 3 + 1], i2 = idx[p * 3 + 2];
  const float w0 = wts[p * 3 + 0], w1 = wts[p * 3 + 1], w2 = wts[p * 3 + 2];
  #pragma unroll
  for (int cc = c; cc < DTF; cc += 128) {
    const float v = w0 * feats_t[(size_t)i0 * DTF + cc]
                  + w1 * feats_t[(size_t)i1 * DTF + cc]
                  + w2 * feats_t[(size_t)i2 * DTF + cc];
    featA[(size_t)p * CIN + DSF + cc] = (_Float16)v;
  }
}

// Assemble a v16h fragment from two contiguous 16-byte halves.
__device__ __forceinline__ v16h ld_frag(const _Float16* lo, const _Float16* hi) {
  const v8h a = *(const v8h*)lo;
  const v8h b = *(const v8h*)hi;
  v16h f;
  #pragma unroll
  for (int i = 0; i < 8; ++i) { f[i] = a[i]; f[i + 8] = b[i]; }
  return f;
}

// ---------------------------------------------------------------------------
// WMMA GEMM:  X[M][NOUT] = A[M][KD] * W[NOUT][KD]^T   (f16 in, fp32 acc)
// One wave -> one 16-row M-tile x NT 16-col N-tiles, register blocked and
// explicitly double buffered: the k+32 fragments are loaded (and kept in a
// second live register set) before the k-step's NT WMMAs issue, so the
// WMMAs run back-to-back while the next loads are in flight.  8 waves/block
// cover 8 M-tiles of the same N-tile group -> W stream is L0/L2 resident.
// A-fragment: row per lane, K runs {8h..8h+7} and {16+8h..}, contiguous b128.
// B-fragment: weight row per lane, K runs {8h..8h+7} and {16+8h..}, contiguous b128.
// Layouts per ISA 7.12.2.
// ---------------------------------------------------------------------------
template <int KD, int NOUT, int NT>
__global__ __launch_bounds__(256) void gemm_wmma_kernel(
    const _Float16* __restrict__ A,
    const _Float16* __restrict__ W,
    float* __restrict__ X) {
  const int wave = threadIdx.x >> 5;
  const int lane = threadIdx.x & 31;
  const int hf   = lane >> 4;                      // 0: lanes 0-15, 1: 16-31
  const int r    = lane & 15;
  const int m0   = (blockIdx.x * 8 + wave) * 16;
  const int n0   = blockIdx.y * (16 * NT);

  const _Float16* __restrict__ arow = A + (size_t)(m0 + r) * KD + 8 * hf;
  const _Float16* brow[NT];
  #pragma unroll
  for (int j = 0; j < NT; ++j)
    brow[j] = W + (size_t)(n0 + 16 * j + r) * KD + 8 * hf;

  v8f acc[NT];
  #pragma unroll
  for (int j = 0; j < NT; ++j) acc[j] = (v8f){};

  // Prologue: fragments for k=0.
  v16h a_cur = ld_frag(arow, arow + 16);
  v16h b_cur[NT];
  #pragma unroll
  for (int j = 0; j < NT; ++j) b_cur[j] = ld_frag(brow[j], brow[j] + 16);

  #pragma unroll
  for (int k0 = 0; k0 < KD; k0 += 32) {
    v16h a_nxt, b_nxt[NT];
    if (k0 + 32 < KD) {                            // issue next-step loads
      if (k0 + 64 < KD)                            // + deep prefetch of the
        __builtin_prefetch(arow + k0 + 64, 0, 3);  //   streaming A operand
      a_nxt = ld_frag(arow + k0 + 32, arow + k0 + 48);
      #pragma unroll
      for (int j = 0; j < NT; ++j)
        b_nxt[j] = ld_frag(brow[j] + k0 + 32, brow[j] + k0 + 48);
    }
    #pragma unroll
    for (int j = 0; j < NT; ++j)                   // NT back-to-back WMMAs
      acc[j] = __builtin_amdgcn_wmma_f32_16x16x32_f16(
          /*neg_a=*/false, a_cur, /*neg_b=*/false, b_cur[j],
          /*c_mod=*/(short)0, acc[j], /*reuse_a=*/false, /*reuse_b=*/false);
    if (k0 + 32 < KD) {
      a_cur = a_nxt;
      #pragma unroll
      for (int j = 0; j < NT; ++j) b_cur[j] = b_nxt[j];
    }
  }

  // D layout: VGPR i -> row m0 + i + 8*hf, col n + r (coalesced per half-wave)
  #pragma unroll
  for (int j = 0; j < NT; ++j) {
    float* out = X + (size_t)(m0 + 8 * hf) * NOUT + n0 + 16 * j + r;
    #pragma unroll
    for (int i = 0; i < 8; ++i) out[(size_t)i * NOUT] = acc[j][i];
  }
}

// ---------------------------------------------------------------------------
// Deterministic per-channel partial sums/sumsq: 128 blocks x C threads,
// each block owns 256 rows; partials summed later in bn_params_kernel.
// ---------------------------------------------------------------------------
template <int C>
__global__ void col_stats_kernel(const float* __restrict__ X,
                                 float* __restrict__ psum,
                                 float* __restrict__ psq) {
  const int c    = threadIdx.x;
  const int row0 = blockIdx.x * 256;
  float s = 0.f, q = 0.f;
  for (int i = 0; i < 256; ++i) {
    const float v = X[(size_t)(row0 + i) * C + c];
    s += v; q += v * v;
  }
  psum[blockIdx.x * C + c] = s;
  psq [blockIdx.x * C + c] = q;
}

// scale/shift with biased variance; conv bias cancels under BN and is omitted.
template <int C>
__global__ void bn_params_kernel(const float* __restrict__ psum,
                                 const float* __restrict__ psq,
                                 const float* __restrict__ gamma,
                                 const float* __restrict__ beta,
                                 float* __restrict__ scale,
                                 float* __restrict__ shift) {
  const int c = threadIdx.x;                       // one block of C threads
  float s = 0.f, q = 0.f;
  for (int i = 0; i < 128; ++i) { s += psum[i * C + c]; q += psq[i * C + c]; }
  const float inv_m = 1.0f / (float)MROW;
  const float mean  = s * inv_m;
  const float var   = q * inv_m - mean * mean;
  const float sc    = gamma[c] * rsqrtf(var + 1e-5f);
  scale[c] = sc;
  shift[c] = beta[c] - mean * sc;
}

// BN + ReLU -> f16 activations for the next WMMA stage.
__global__ void bn_relu_to_f16_kernel(const float* __restrict__ X0,
                                      const float* __restrict__ scale,
                                      const float* __restrict__ shift,
                                      _Float16* __restrict__ A1) {
  const size_t e = (size_t)blockIdx.x * 256 + threadIdx.x;   // MROW*C0 elems
  const int    c = (int)(e & (C0F - 1));
  float v = X0[e] * scale[c] + shift[c];
  v = v > 0.f ? v : 0.f;
  A1[e] = (_Float16)v;
}

// In-place BN + ReLU on the final fp32 output buffer.
__global__ void bn_relu_final_kernel(float* __restrict__ Y,
                                     const float* __restrict__ scale,
                                     const float* __restrict__ shift) {
  const size_t e = (size_t)blockIdx.x * 256 + threadIdx.x;   // MROW*C1 elems
  const int    c = (int)(e & (C1F - 1));
  float v = Y[e] * scale[c] + shift[c];
  Y[e] = v > 0.f ? v : 0.f;
}

// ---------------------------------------------------------------------------
extern "C" void kernel_launch(void* const* d_in, const int* in_sizes, int n_in,
                              void* d_out, int out_size, void* d_ws, size_t ws_size,
                              hipStream_t stream) {
  (void)in_sizes; (void)n_in; (void)out_size; (void)ws_size;

  const float* xyz_s   = (const float*)d_in[0];
  const float* xyz_t   = (const float*)d_in[1];
  const float* feats_s = (const float*)d_in[2];
  const float* feats_t = (const float*)d_in[3];
  const float* w0      = (const float*)d_in[4];
  // d_in[5] = b0 : cancels exactly under training-mode BN -> unused
  const float* gamma0  = (const float*)d_in[6];
  const float* beta0   = (const float*)d_in[7];
  const float* w1      = (const float*)d_in[8];
  // d_in[9] = b1 : cancels under BN -> unused
  const float* gamma1  = (const float*)d_in[10];
  const float* beta1   = (const float*)d_in[11];
  float* out = (float*)d_out;

  char* ws = (char*)d_ws;
  _Float16* featA = (_Float16*)(ws + OFF_FEAT);
  _Float16* wb0   = (_Float16*)(ws + OFF_WB0);
  _Float16* wb1   = (_Float16*)(ws + OFF_WB1);
  float*    x0    = (float*)   (ws + OFF_X0);
  _Float16* a1    = (_Float16*)(ws + OFF_A1);
  int*      idx   = (int*)     (ws + OFF_IDX);
  float*    wts   = (float*)   (ws + OFF_WTS);
  float*    st    = (float*)   (ws + OFF_STATS);

  // 1. weights -> f16 (L2 resident afterwards)
  cvt_weights_kernel<<<512, 256, 0, stream>>>(w0, w1, wb0, wb1);

  // 2. 3-NN (LDS-tiled brute force)
  knn_kernel<<<MROW / 256, 256, 0, stream>>>(xyz_s, xyz_t, idx, wts);

  // 3. gather + interp + concat -> f16 features [MROW][384]
  build_features_kernel<<<MROW, 128, 0, stream>>>(feats_s, feats_t, idx, wts, featA);

  // 4. GEMM1 (384 -> 256): 4-way N blocking, grid (256, 4)
  gemm_wmma_kernel<CIN, C0F, 4>
      <<<dim3(MROW / 128, C0F / 64), 256, 0, stream>>>(featA, wb0, x0);

  // 5. BN0 stats + params
  col_stats_kernel<C0F><<<128, C0F, 0, stream>>>(x0, st + ST_PSUM0, st + ST_PSQ0);
  bn_params_kernel<C0F><<<1, C0F, 0, stream>>>(st + ST_PSUM0, st + ST_PSQ0,
                                               gamma0, beta0,
                                               st + ST_SCALE0, st + ST_SHIFT0);

  // 6. BN0 + ReLU -> f16
  bn_relu_to_f16_kernel<<<(MROW * C0F) / 256, 256, 0, stream>>>(
      x0, st + ST_SCALE0, st + ST_SHIFT0, a1);

  // 7. GEMM2 (256 -> 128): 4-way N blocking, grid (256, 2); pre-BN -> d_out
  gemm_wmma_kernel<C0F, C1F, 4>
      <<<dim3(MROW / 128, C1F / 64), 256, 0, stream>>>(a1, wb1, out);

  // 8. BN1 stats + params
  col_stats_kernel<C1F><<<128, C1F, 0, stream>>>(out, st + ST_PSUM1, st + ST_PSQ1);
  bn_params_kernel<C1F><<<1, C1F, 0, stream>>>(st + ST_PSUM1, st + ST_PSQ1,
                                               gamma1, beta1,
                                               st + ST_SCALE1, st + ST_SHIFT1);

  // 9. BN1 + ReLU in place
  bn_relu_final_kernel<<<(MROW * C1F) / 256, 256, 0, stream>>>(
      out, st + ST_SCALE1, st + ST_SHIFT1);
}
